// DynamicMultiheadAttention_48850958025159
// MI455X (gfx1250) — matrix operand order repaired and checked by reading, not compiled, not run on hardware
//
#include <hip/hip_runtime.h>
#include <hip/hip_bf16.h>

namespace {
constexpr int kN  = 8192;
constexpr int kD  = 512;
constexpr int kG  = 64;
constexpr int kNP = 9248;       // permuted rows, each group padded to 16, + slack (9248 = 578*16)
constexpr int kMetaInts = 192;  // meta: cnt[64] @0, poff[65] @64, inv[kNP] @192
}

typedef _Float16 v16h __attribute__((ext_vector_type(16)));
typedef float    v8f  __attribute__((ext_vector_type(8)));

#define WMMA16(a, b, c) \
  __builtin_amdgcn_wmma_f32_16x16x32_f16(false, (a), false, (b), (short)0, (c), false, false)

// A operand, 16-bit, 16x32: lane half 0 -> K 0..15, half 1 -> K 16..31.
// Caller passes p = rowBase + k0 + hi*16; 16 contiguous halves.
__device__ __forceinline__ v16h ld_a_h(const _Float16* p) {
  v16h r;
#pragma unroll
  for (int i = 0; i < 16; ++i) r[i] = p[i];
  return r;
}
// B operand, 16-bit, 32x16 stored as N x K (row-major BT): lane = col n, half picks K 0-15 / 16-31.
// Caller passes p = BT + n*ld + k0 + hi*16; 16 contiguous halves (32B aligned).
__device__ __forceinline__ v16h ld_b_h(const _Float16* p) {
  v16h r;
#pragma unroll
  for (int i = 0; i < 16; ++i) r[i] = p[i];
  return r;
}

// ---------------- prep kernels ----------------

__global__ void prep_wt_kernel(const float* __restrict__ Wq, const float* __restrict__ Wk,
                               const float* __restrict__ Wv, const float* __restrict__ Wo,
                               _Float16* __restrict__ WT) {
  const int t = blockIdx.x * 256 + threadIdx.x;  // [0, 4*D*D)
  const int z = t >> 18;                         // D*D == 2^18
  const int e = t & (kD * kD - 1);
  const int n = e >> 9, k = e & (kD - 1);
  const float* W = (z == 0) ? Wq : (z == 1) ? Wk : (z == 2) ? Wv : Wo;
  WT[t] = (_Float16)W[k * kD + n];               // WT[z][n][k] = W[k][n]
}

__global__ void conv_x_kernel(const float* __restrict__ x, _Float16* __restrict__ xh) {
  const int t = blockIdx.x * 256 + threadIdx.x;  // [0, N*D)
  xh[t] = (_Float16)x[t];
}

__global__ void zero_meta_kernel(int* __restrict__ meta) {
  const int t = blockIdx.x * 256 + threadIdx.x;
  if (t < kMetaInts + kNP) meta[t] = (t >= kMetaInts) ? -1 : 0;
}

__global__ void count_kernel(const int* __restrict__ labels, int* __restrict__ meta) {
  const int i = blockIdx.x * 256 + threadIdx.x;
  if (i < kN) { const int l = labels[i]; if (l >= 0) atomicAdd(&meta[l], 1); }
}

__global__ void scan_kernel(int* __restrict__ meta) {
  int a = 0;
  for (int g = 0; g < kG; ++g) { meta[64 + g] = a; a += (meta[g] + 15) & ~15; }
  meta[64 + kG] = a;
}

// deterministic rank within label (no FP-order nondeterminism)
__global__ void build_inv_kernel(const int* __restrict__ labels, int* __restrict__ meta) {
  const int i = blockIdx.x * 256 + threadIdx.x;
  if (i >= kN) return;
  const int l = labels[i];
  if (l < 0) return;
  int rank = 0;
  for (int j = 0; j < i; ++j) rank += (labels[j] == l) ? 1 : 0;
  meta[kMetaInts + meta[64 + l] + rank] = i;
}

__global__ void init_out_kernel(float* __restrict__ out, const float* __restrict__ bo) {
  const int t = blockIdx.x * 256 + threadIdx.x;
  out[t] = bo[t & (kD - 1)];   // invalid rows end up as 0 @ Wo + bo = bo
}

// ---------------- QKV projection (gathered) ----------------
// grid (kNP/16, D/64, 3), block 128 (4 waves, each a 16x16 out tile)

__global__ void __launch_bounds__(128)
qkv_proj_kernel(const _Float16* __restrict__ xh, const _Float16* __restrict__ WT,
                const float* __restrict__ bq, const float* __restrict__ bk,
                const float* __restrict__ bv, const int* __restrict__ inv,
                _Float16* __restrict__ qg, _Float16* __restrict__ kg,
                _Float16* __restrict__ vT) {
  const int lane = threadIdx.x & 31;
  const int wave = threadIdx.x >> 5;
  const int mi = lane & 15, hi = lane >> 4;
  const int tile = blockIdx.x;
  const int n0 = blockIdx.y * 64 + wave * 16;
  const int z = blockIdx.z;
  const _Float16* Wt = WT + (size_t)z * kD * kD;
  const float* bias = (z == 0) ? bq : (z == 1) ? bk : bv;

  int src = inv[tile * 16 + mi];
  if (src < 0) src = 0;                       // pad rows -> finite values (never stored back)
  const _Float16* arow = xh + (size_t)src * kD + hi * 16;
  const _Float16* brow = Wt + (size_t)(n0 + mi) * kD + hi * 16;

  // two independent accumulator chains to avoid WMMA->WMMA hazard NOPs
  v8f acc0 = {}, acc1 = {};
#pragma unroll
  for (int k0 = 0; k0 < kD; k0 += 64) {
    acc0 = WMMA16(ld_a_h(arow + k0),      ld_b_h(brow + k0),      acc0);
    acc1 = WMMA16(ld_a_h(arow + k0 + 32), ld_b_h(brow + k0 + 32), acc1);
  }
  const v8f acc = acc0 + acc1;

  const float bn = bias[n0 + mi];
  if (z == 2) {
#pragma unroll
    for (int r = 0; r < 8; ++r) {
      const int p = tile * 16 + r + hi * 8;
      vT[(size_t)(n0 + mi) * kNP + p] = (_Float16)(acc[r] + bn);   // v stored transposed
    }
  } else {
    _Float16* dst = (z == 0) ? qg : kg;
#pragma unroll
    for (int r = 0; r < 8; ++r) {
      const int p = tile * 16 + r + hi * 8;
      dst[(size_t)p * kD + n0 + mi] = (_Float16)(acc[r] + bn);
    }
  }
}

// ---------------- block-diagonal flash attention ----------------
// grid (64 groups, 32 query tiles), block 256 (8 waves). Wave w owns D-slice [64w, 64w+64).

__global__ void __launch_bounds__(256)
attn_kernel(const _Float16* __restrict__ qg, const _Float16* __restrict__ kg,
            const _Float16* __restrict__ vT, const int* __restrict__ meta,
            _Float16* __restrict__ attg) {
  __shared__ _Float16 sQ[16 * kD];
  __shared__ _Float16 sP[16 * 32];
  const int g = blockIdx.x;
  const int t = blockIdx.y;
  const int rows = meta[g];
  if (t * 16 >= rows) return;
  const int base = meta[64 + g];
  const int qbase = base + t * 16;

  {  // stage 16x512 f16 Q tile into LDS (1024 x uint4)
    const uint4* s4 = (const uint4*)(qg + (size_t)qbase * kD);
    uint4* d4 = (uint4*)sQ;
#pragma unroll
    for (int i = 0; i < 4; ++i) d4[threadIdx.x + 256 * i] = s4[threadIdx.x + 256 * i];
  }
  __syncthreads();

  const int lane = threadIdx.x & 31;
  const int wave = threadIdx.x >> 5;
  const int mi = lane & 15, hi = lane >> 4;
  const int nw = wave * 64;
  const float scale = 0.044194173824159216f;  // 1/sqrt(512)

  v8f o0 = {}, o1 = {}, o2 = {}, o3 = {};
  float mrun[8], lrun[8];
#pragma unroll
  for (int r = 0; r < 8; ++r) { mrun[r] = -1e30f; lrun[r] = 0.0f; }

  for (int ks = 0; ks < rows; ks += 32) {
    // S[16x32] = scale * Q @ K^T (all waves compute identical copy);
    // four independent accumulator chains hide WMMA RAW latency
    v8f s0a = {}, s0b = {}, s1a = {}, s1b = {};
    const _Float16* qrow = sQ + mi * kD + hi * 16;
    const _Float16* k0p = kg + (size_t)(base + ks + mi) * kD + hi * 16;
    const _Float16* k1p = k0p + (size_t)16 * kD;
    if (ks + 32 < rows)  // stream next key chunk into cache (global_prefetch_b8)
      __builtin_prefetch(k0p + (size_t)32 * kD, 0, 3);
#pragma unroll
    for (int k0 = 0; k0 < kD; k0 += 64) {
      v16h a0 = ld_a_h(qrow + k0);
      v16h a1 = ld_a_h(qrow + k0 + 32);
      s0a = WMMA16(a0, ld_b_h(k0p + k0),      s0a);
      s1a = WMMA16(a0, ld_b_h(k1p + k0),      s1a);
      s0b = WMMA16(a1, ld_b_h(k0p + k0 + 32), s0b);
      s1b = WMMA16(a1, ld_b_h(k1p + k0 + 32), s1b);
    }
    v8f s0 = s0a + s0b;
    v8f s1 = s1a + s1b;

    const bool v0c = (ks + mi) < rows;        // column mask (same col for all 8 rows of a lane)
    const bool v1c = (ks + 16 + mi) < rows;
#pragma unroll
    for (int r = 0; r < 8; ++r) {
      s0[r] = v0c ? s0[r] * scale : -1e30f;
      s1[r] = v1c ? s1[r] * scale : -1e30f;
      float mx = fmaxf(s0[r], s1[r]);
#pragma unroll
      for (int o = 1; o < 16; o <<= 1) mx = fmaxf(mx, __shfl_xor(mx, o, 32));
      const float mnew = fmaxf(mrun[r], mx);
      const float rs = __expf(mrun[r] - mnew);
      mrun[r] = mnew;
      s0[r] = __expf(s0[r] - mnew);
      s1[r] = __expf(s1[r] - mnew);
      float sm = s0[r] + s1[r];
#pragma unroll
      for (int o = 1; o < 16; o <<= 1) sm += __shfl_xor(sm, o, 32);
      lrun[r] = lrun[r] * rs + sm;
      o0[r] *= rs; o1[r] *= rs; o2[r] *= rs; o3[r] *= rs;
    }
    __syncthreads();
    if (wave == 0) {  // re-lay-out P from C layout into A-operand source tile
#pragma unroll
      for (int r = 0; r < 8; ++r) {
        const int row = r + hi * 8;
        sP[row * 32 + mi]      = (_Float16)s0[r];
        sP[row * 32 + 16 + mi] = (_Float16)s1[r];
      }
    }
    __syncthreads();
    // O += P @ V ; B operand from vT (N x K contiguous, 32B aligned since base%16==0)
    v16h pa = ld_a_h(sP + mi * 32 + hi * 16);
    const _Float16* vb = vT + (size_t)base + ks + hi * 16;
    o0 = WMMA16(pa, ld_b_h(vb + (size_t)(nw +  0 + mi) * kNP), o0);
    o1 = WMMA16(pa, ld_b_h(vb + (size_t)(nw + 16 + mi) * kNP), o1);
    o2 = WMMA16(pa, ld_b_h(vb + (size_t)(nw + 32 + mi) * kNP), o2);
    o3 = WMMA16(pa, ld_b_h(vb + (size_t)(nw + 48 + mi) * kNP), o3);
  }
#pragma unroll
  for (int r = 0; r < 8; ++r) {
    const int qrow = t * 16 + r + hi * 8;
    if (qrow < rows) {
      const float il = 1.0f / fmaxf(lrun[r], 1e-30f);
      _Float16* dst = attg + (size_t)(base + qrow) * kD + nw + mi;
      dst[0]  = (_Float16)(o0[r] * il);
      dst[16] = (_Float16)(o1[r] * il);
      dst[32] = (_Float16)(o2[r] * il);
      dst[48] = (_Float16)(o3[r] * il);
    }
  }
}

// ---------------- output projection + scatter ----------------
// grid (kNP/16, D/64), block 128

__global__ void __launch_bounds__(128)
out_proj_kernel(const _Float16* __restrict__ attg, const _Float16* __restrict__ WoT,
                const float* __restrict__ bo, const int* __restrict__ inv,
                float* __restrict__ out) {
  const int lane = threadIdx.x & 31;
  const int wave = threadIdx.x >> 5;
  const int mi = lane & 15, hi = lane >> 4;
  const int tile = blockIdx.x;
  const int n0 = blockIdx.y * 64 + wave * 16;

  const _Float16* arow = attg + (size_t)(tile * 16 + mi) * kD + hi * 16;
  const _Float16* brow = WoT + (size_t)(n0 + mi) * kD + hi * 16;
  v8f acc0 = {}, acc1 = {};
#pragma unroll
  for (int k0 = 0; k0 < kD; k0 += 64) {
    acc0 = WMMA16(ld_a_h(arow + k0),      ld_b_h(brow + k0),      acc0);
    acc1 = WMMA16(ld_a_h(arow + k0 + 32), ld_b_h(brow + k0 + 32), acc1);
  }
  const v8f acc = acc0 + acc1;

  const float bn = bo[n0 + mi];
#pragma unroll
  for (int r = 0; r < 8; ++r) {
    const int p = tile * 16 + r + hi * 8;
    const int row = inv[p];
    if (row >= 0) out[(size_t)row * kD + n0 + mi] = acc[r] + bn;
  }
}

// ---------------- launch ----------------

extern "C" void kernel_launch(void* const* d_in, const int* in_sizes, int n_in,
                              void* d_out, int out_size, void* d_ws, size_t ws_size,
                              hipStream_t stream) {
  (void)in_sizes; (void)n_in; (void)out_size; (void)ws_size;
  const float* x      = (const float*)d_in[0];
  const int*   labels = (const int*)d_in[1];
  const float* Wq = (const float*)d_in[2];
  const float* bq = (const float*)d_in[3];
  const float* Wk = (const float*)d_in[4];
  const float* bk = (const float*)d_in[5];
  const float* Wv = (const float*)d_in[6];
  const float* bv = (const float*)d_in[7];
  const float* Wo = (const float*)d_in[8];
  const float* bo = (const float*)d_in[9];
  float* out = (float*)d_out;

  _Float16* WT   = (_Float16*)d_ws;                  // 4 transposed f16 weight mats
  _Float16* xh   = WT + (size_t)4 * kD * kD;         // x converted to f16 once
  _Float16* qg   = xh + (size_t)kN * kD;
  _Float16* kg   = qg + (size_t)kNP * kD;
  _Float16* vT   = kg + (size_t)kNP * kD;
  _Float16* attg = vT + (size_t)kNP * kD;
  int* meta = (int*)(attg + (size_t)kNP * kD);       // ~48 MB total workspace

  prep_wt_kernel<<<(4 * kD * kD) / 256, 256, 0, stream>>>(Wq, Wk, Wv, Wo, WT);
  conv_x_kernel<<<(kN * kD) / 256, 256, 0, stream>>>(x, xh);
  zero_meta_kernel<<<(kMetaInts + kNP + 255) / 256, 256, 0, stream>>>(meta);
  count_kernel<<<kN / 256, 256, 0, stream>>>(labels, meta);
  scan_kernel<<<1, 1, 0, stream>>>(meta);
  build_inv_kernel<<<kN / 256, 256, 0, stream>>>(labels, meta);
  qkv_proj_kernel<<<dim3(kNP / 16, kD / 64, 3), 128, 0, stream>>>(
      xh, WT, bq, bk, bv, meta + kMetaInts, qg, kg, vT);
  attn_kernel<<<dim3(kG, 32), 256, 0, stream>>>(qg, kg, vT, meta, attg);
  init_out_kernel<<<(kN * kD) / 256, 256, 0, stream>>>(out, bo);
  out_proj_kernel<<<dim3(kNP / 16, kD / 64), 128, 0, stream>>>(
      attg, WT + (size_t)3 * kD * kD, bo, meta + kMetaInts, out);
}
